// GraphEnsemble_3410204033759
// MI455X (gfx1250) — matrix operand order repaired and checked by reading, not compiled, not run on hardware
//
#include <hip/hip_runtime.h>
#include <hip/hip_bf16.h>
#include <math.h>

// ---------------- problem constants ----------------
#define NNODES 50000
#define NEDGES 800000
#define FIN    128
#define HID    64
#define NCLS   40
#define CPAD   48      // NCLS padded to 3x16 WMMA tiles

typedef __attribute__((ext_vector_type(16))) _Float16 v16h;
typedef __attribute__((ext_vector_type(8)))  _Float16 v8h;
typedef __attribute__((ext_vector_type(8)))  float    v8f;
typedef _Float16 h16;

__device__ __forceinline__ float gelu_exact(float v) {
    return 0.5f * v * (1.0f + erff(v * 0.70710678118654752f));
}

__device__ __forceinline__ void atomicMaxFloat(float* addr, float val) {
    int old = __float_as_int(*addr);
    while (__int_as_float(old) < val) {
        int prev = atomicCAS((int*)addr, old, __float_as_int(val));
        if (prev == old) break;
        old = prev;
    }
}

// ---------------- small utility kernels ----------------
__global__ void k_fill_f32(float* p, float v, int n) {
    int i = blockIdx.x * blockDim.x + threadIdx.x;
    if (i < n) p[i] = v;
}

__global__ void k_f32_to_f16(const float* __restrict__ s, h16* __restrict__ d, int n) {
    int i = blockIdx.x * blockDim.x + threadIdx.x;
    if (i < n) d[i] = (h16)s[i];
}

// Pack fp32 weight W[K][N] into the WMMA B-operand per-lane layout (f16), with
// zero padding to ntiles*16 columns.  Output element (ks, nt, lane, j) holds
// B[k = ks*32 + (lane>>4)*16 + j][n = nt*16 + (lane&15)] so the GEMM kernel can
// fetch each lane's 16 values with one aligned 32-byte load.
__global__ void k_pack_b(const float* __restrict__ W, h16* __restrict__ pb,
                         int K, int N, int ntiles) {
    int idx = blockIdx.x * blockDim.x + threadIdx.x;
    int total = (K >> 5) * ntiles * 32 * 16;
    if (idx >= total) return;
    int j  = idx & 15;
    int l  = (idx >> 4) & 31;
    int nt = (idx >> 9) % ntiles;
    int ks = idx / (512 * ntiles);
    int n  = nt * 16 + (l & 15);
    int k  = ks * 32 + (l >> 4) * 16 + j;
    pb[idx] = (n < N) ? (h16)W[(size_t)k * N + n] : (h16)0.0f;
}

// ---------------- WMMA GEMM: one wave computes a 16 x (NT*16) row slab -------
// A[M x K] row-major f16 (lda = K), Bp = packed B (see k_pack_b), C fp32.
// K multiple of 32, M multiple of 16.  A fragment loaded once per k-step and
// reused across NT WMMAs.
template<int NT>
__global__ void k_wmma_gemm(const h16* __restrict__ A, int lda,
                            const h16* __restrict__ Bp,
                            float* __restrict__ C, int ldc,
                            int mtiles, int K) {
    int wave = (blockIdx.x * blockDim.x + threadIdx.x) >> 5;
    if (wave >= mtiles) return;                 // whole-wave uniform exit
    int lane = threadIdx.x & 31;
    int row0 = wave << 4;
    int m  = lane & 15;                         // A row / C col within tile
    int hh = lane >> 4;                         // lane-half selector

    v8f acc[NT];
    #pragma unroll
    for (int t = 0; t < NT; ++t) acc[t] = (v8f){};

    const v16h* bp = (const v16h*)Bp;
    const int ksteps = K >> 5;
    #pragma unroll 4
    for (int ks = 0; ks < ksteps; ++ks) {
        // A 16x32 f16 layout: lanes 0-15 -> K=+0..15 ; 16-31 -> +16..31 (matches packed B)
        const v8h* ap = (const v8h*)(A + (size_t)(row0 + m) * lda + ks * 32 + hh * 16);
        v8h a0 = ap[0];
        v8h a1 = ap[1];
        v16h a = __builtin_shufflevector(a0, a1, 0, 1, 2, 3, 4, 5, 6, 7,
                                                 8, 9, 10, 11, 12, 13, 14, 15);
        #pragma unroll
        for (int t = 0; t < NT; ++t) {
            v16h b = bp[(size_t)(ks * NT + t) * 32 + lane];   // one 32B aligned load
            acc[t] = __builtin_amdgcn_wmma_f32_16x16x32_f16(false, a, false, b,
                                                            (short)0, acc[t], false, false);
        }
    }
    // C/D layout: VGPR r, lane -> M = r + 8*hh, N = lane&15
    #pragma unroll
    for (int t = 0; t < NT; ++t) {
        #pragma unroll
        for (int r = 0; r < 8; ++r)
            C[(size_t)(row0 + r + hh * 8) * ldc + t * 16 + m] = acc[t][r];
    }
}

// ---------------- edge scatter kernels ----------------
// SAGE: agg[dst] += x[src] (128 f), cnt[dst] += 1.  wave per edge.
__global__ void k_sage_scatter(const float* __restrict__ x, const long long* __restrict__ ei,
                               float* __restrict__ agg, float* __restrict__ cnt) {
    int t = blockIdx.x * blockDim.x + threadIdx.x;
    int e = t >> 5;
    if (e >= NEDGES) return;
    int lane = t & 31;
    int s = (int)ei[e], d = (int)ei[NEDGES + e];
    const float* xs = x + (size_t)s * FIN;
    float* ad = agg + (size_t)d * FIN;
    #pragma unroll
    for (int j = 0; j < 4; ++j)
        atomicAdd(&ad[lane + 32 * j], xs[lane + 32 * j]);
    if (lane == 0) atomicAdd(&cnt[d], 1.0f);
}

__global__ void k_sage_norm(const float* __restrict__ agg, const float* __restrict__ cnt,
                            h16* __restrict__ aggh) {
    int t = blockIdx.x * blockDim.x + threadIdx.x;
    if (t >= NNODES * FIN) return;
    float c = cnt[t / FIN];
    c = c > 1.0f ? c : 1.0f;
    aggh[t] = (h16)(agg[t] / c);
}

__global__ void k_deg(const long long* __restrict__ ei, float* __restrict__ deg) {
    int e = blockIdx.x * blockDim.x + threadIdx.x;
    if (e < NEDGES) atomicAdd(&deg[(int)ei[NEDGES + e]], 1.0f);
}

__global__ void k_dinv(const float* __restrict__ deg, float* __restrict__ dinv) {
    int i = blockIdx.x * blockDim.x + threadIdx.x;
    if (i < NNODES) dinv[i] = rsqrtf(deg[i]);   // deg >= 1 (self loop folded)
}

// GCN: acc[dst] += h[src] * dinv[src]*dinv[dst].  wave per edge, 64 feats.
__global__ void k_gcn_scatter(const float* __restrict__ h, const long long* __restrict__ ei,
                              const float* __restrict__ dinv, float* __restrict__ acc) {
    int t = blockIdx.x * blockDim.x + threadIdx.x;
    int e = t >> 5;
    if (e >= NEDGES) return;
    int lane = t & 31;
    int s = (int)ei[e], d = (int)ei[NEDGES + e];
    float nrm = dinv[s] * dinv[d];
    const float* hs = h + (size_t)s * HID;
    float* ad = acc + (size_t)d * HID;
    atomicAdd(&ad[lane],      hs[lane]      * nrm);
    atomicAdd(&ad[lane + 32], hs[lane + 32] * nrm);
}

// GAT per-node attention scalars: a_s[i]=h[i].att_src, a_d[i]=h[i].att_dst
__global__ void k_gat_pre(const float* __restrict__ h, const float* __restrict__ asrc,
                          const float* __restrict__ adst,
                          float* __restrict__ a_s, float* __restrict__ a_d) {
    int t = blockIdx.x * blockDim.x + threadIdx.x;
    int i = t >> 5;
    if (i >= NNODES) return;
    int lane = t & 31;
    const float* hp = h + (size_t)i * HID;
    float vs = hp[lane] * asrc[lane] + hp[lane + 32] * asrc[lane + 32];
    float vd = hp[lane] * adst[lane] + hp[lane + 32] * adst[lane + 32];
    #pragma unroll
    for (int off = 16; off > 0; off >>= 1) {
        vs += __shfl_xor(vs, off, 32);
        vd += __shfl_xor(vd, off, 32);
    }
    if (lane == 0) { a_s[i] = vs; a_d[i] = vd; }
}

__device__ __forceinline__ void gat_edge(int idx, const long long* ei, int* s, int* d) {
    if (idx < NEDGES) { *s = (int)ei[idx]; *d = (int)ei[NEDGES + idx]; }
    else              { *s = *d = idx - NEDGES; }   // self loops appended
}

__global__ void k_gat_max(const long long* __restrict__ ei, const float* __restrict__ a_s,
                          const float* __restrict__ a_d, float* __restrict__ gmax) {
    int idx = blockIdx.x * blockDim.x + threadIdx.x;
    if (idx >= NEDGES + NNODES) return;
    int s, d; gat_edge(idx, ei, &s, &d);
    float e = a_s[s] + a_d[d];
    e = e > 0.0f ? e : 0.2f * e;
    atomicMaxFloat(&gmax[d], e);
}

__global__ void k_gat_sum(const long long* __restrict__ ei, const float* __restrict__ a_s,
                          const float* __restrict__ a_d, const float* __restrict__ gmax,
                          float* __restrict__ gsum) {
    int idx = blockIdx.x * blockDim.x + threadIdx.x;
    if (idx >= NEDGES + NNODES) return;
    int s, d; gat_edge(idx, ei, &s, &d);
    float e = a_s[s] + a_d[d];
    e = e > 0.0f ? e : 0.2f * e;
    atomicAdd(&gsum[d], __expf(e - gmax[d]));
}

__global__ void k_gat_scatter(const long long* __restrict__ ei, const float* __restrict__ a_s,
                              const float* __restrict__ a_d, const float* __restrict__ gmax,
                              const float* __restrict__ gsum, const float* __restrict__ h,
                              float* __restrict__ acc) {
    int t = blockIdx.x * blockDim.x + threadIdx.x;
    int idx = t >> 5;
    if (idx >= NEDGES + NNODES) return;
    int lane = t & 31;
    int s, d; gat_edge(idx, ei, &s, &d);
    float e = a_s[s] + a_d[d];
    e = e > 0.0f ? e : 0.2f * e;
    float alpha = __expf(e - gmax[d]) / (gsum[d] + 1e-16f);
    const float* hs = h + (size_t)s * HID;
    float* ad = acc + (size_t)d * HID;
    atomicAdd(&ad[lane],      alpha * hs[lane]);
    atomicAdd(&ad[lane + 32], alpha * hs[lane + 32]);
}

// ---------------- ensemble: gelu + weighted sum, emit fp32 + f16 ----------------
__global__ void k_ens(const float* __restrict__ gcnacc, const float* __restrict__ hgcn,
                      const float* __restrict__ dinv, const float* __restrict__ bgcn,
                      const float* __restrict__ hsl, const float* __restrict__ bl,
                      const float* __restrict__ hsr,
                      const float* __restrict__ gatacc, const float* __restrict__ bgat,
                      const float* __restrict__ w,
                      float* __restrict__ ens, h16* __restrict__ ensh) {
    int t = blockIdx.x * blockDim.x + threadIdx.x;
    if (t >= NNODES * HID) return;
    int i = t / HID, f = t % HID;
    float di = dinv[i];
    float gc = gcnacc[t] + hgcn[t] * di * di + bgcn[f];   // fold self-loop here
    float sg = hsl[t] + bl[f] + hsr[t];
    float ga = gatacc[t] + bgat[f];
    float e = gelu_exact(gc) * w[0] + gelu_exact(sg) * w[1] + gelu_exact(ga) * w[2];
    ens[t] = e;
    ensh[t] = (h16)e;
}

// ---------------- transformer conv ----------------
__global__ void k_bias4(float* __restrict__ q, float* __restrict__ k,
                        float* __restrict__ v, float* __restrict__ s,
                        const float* __restrict__ bq, const float* __restrict__ bk,
                        const float* __restrict__ bv, const float* __restrict__ bs) {
    int t = blockIdx.x * blockDim.x + threadIdx.x;
    if (t >= NNODES * CPAD) return;
    int c = t % CPAD;
    if (c >= NCLS) return;          // pad cols stay 0
    q[t] += bq[c]; k[t] += bk[c]; v[t] += bv[c]; s[t] += bs[c];
}

__global__ void k_tscore(const float* __restrict__ q, const float* __restrict__ k,
                         const long long* __restrict__ ei,
                         float* __restrict__ tscore, float* __restrict__ tmax) {
    int t = blockIdx.x * blockDim.x + threadIdx.x;
    int e = t >> 5;
    if (e >= NEDGES) return;
    int lane = t & 31;
    int s = (int)ei[e], d = (int)ei[NEDGES + e];
    const float* qp = q + (size_t)d * CPAD;
    const float* kp = k + (size_t)s * CPAD;
    float p = qp[lane] * kp[lane];
    if (lane < 16) p += qp[lane + 32] * kp[lane + 32];   // pad cols are zero
    #pragma unroll
    for (int off = 16; off > 0; off >>= 1) p += __shfl_xor(p, off, 32);
    if (lane == 0) {
        float sc = p * 0.15811388300841898f;  // 1/sqrt(40)
        tscore[e] = sc;
        atomicMaxFloat(&tmax[d], sc);
    }
}

__global__ void k_tsum(const long long* __restrict__ ei, float* __restrict__ tscore,
                       const float* __restrict__ tmax, float* __restrict__ tsum) {
    int e = blockIdx.x * blockDim.x + threadIdx.x;
    if (e >= NEDGES) return;
    int d = (int)ei[NEDGES + e];
    float ex = __expf(tscore[e] - tmax[d]);
    tscore[e] = ex;                 // overwrite with numerator
    atomicAdd(&tsum[d], ex);
}

__global__ void k_tscatter(const long long* __restrict__ ei, const float* __restrict__ tscore,
                           const float* __restrict__ tsum, const float* __restrict__ v,
                           float* __restrict__ tout) {
    int t = blockIdx.x * blockDim.x + threadIdx.x;
    int e = t >> 5;
    if (e >= NEDGES) return;
    int lane = t & 31;
    int s = (int)ei[e], d = (int)ei[NEDGES + e];
    float alpha = tscore[e] / (tsum[d] + 1e-16f);
    const float* vs = v + (size_t)s * CPAD;
    float* od = tout + (size_t)d * CPAD;
    atomicAdd(&od[lane], alpha * vs[lane]);
    if (lane < 16) atomicAdd(&od[lane + 32], alpha * vs[lane + 32]);
}

// ---------------- final: + skip, log_softmax over 40 classes ----------------
__global__ void k_final(const float* __restrict__ tout, const float* __restrict__ skp,
                        float* __restrict__ out) {
    int t = blockIdx.x * blockDim.x + threadIdx.x;
    int i = t >> 5;
    if (i >= NNODES) return;
    int lane = t & 31;
    const float* tp = tout + (size_t)i * CPAD;
    const float* sp = skp + (size_t)i * CPAD;
    float o0 = tp[lane] + sp[lane];                                   // c = lane (<40)
    float o1 = (lane < 8) ? (tp[lane + 32] + sp[lane + 32]) : -INFINITY;
    float mx = fmaxf(o0, o1);
    #pragma unroll
    for (int off = 16; off > 0; off >>= 1) mx = fmaxf(mx, __shfl_xor(mx, off, 32));
    float se = __expf(o0 - mx) + ((lane < 8) ? __expf(o1 - mx) : 0.0f);
    #pragma unroll
    for (int off = 16; off > 0; off >>= 1) se += __shfl_xor(se, off, 32);
    float lse = mx + __logf(se);
    out[(size_t)i * NCLS + lane] = o0 - lse;
    if (lane < 8) out[(size_t)i * NCLS + lane + 32] = o1 - lse;
}

// ---------------- host side ----------------
static inline int gblk(long long threads) { return (int)((threads + 255) / 256); }

extern "C" void kernel_launch(void* const* d_in, const int* in_sizes, int n_in,
                              void* d_out, int out_size, void* d_ws, size_t ws_size,
                              hipStream_t stream) {
    const float*     x     = (const float*)d_in[0];
    const long long* ei    = (const long long*)d_in[1];
    const float*     w     = (const float*)d_in[2];
    const float* Wgcn  = (const float*)d_in[3];  const float* bgcn = (const float*)d_in[4];
    const float* Wl    = (const float*)d_in[5];  const float* bl   = (const float*)d_in[6];
    const float* Wr    = (const float*)d_in[7];
    const float* Wgat  = (const float*)d_in[8];  const float* asrc = (const float*)d_in[9];
    const float* adst  = (const float*)d_in[10]; const float* bgat = (const float*)d_in[11];
    const float* Wq = (const float*)d_in[12]; const float* bq = (const float*)d_in[13];
    const float* Wk = (const float*)d_in[14]; const float* bk = (const float*)d_in[15];
    const float* Wv = (const float*)d_in[16]; const float* bv = (const float*)d_in[17];
    const float* Ws = (const float*)d_in[18]; const float* bs = (const float*)d_in[19];
    float* out = (float*)d_out;

    // workspace carving
    char* ws = (char*)d_ws;
    size_t off = 0;
    auto carve = [&](size_t bytes) -> void* {
        void* p = ws + off;
        off = (off + bytes + 255) & ~(size_t)255;
        return p;
    };
    h16*   xh     = (h16*)carve((size_t)NNODES * FIN * 2);
    h16*   aggh   = (h16*)carve((size_t)NNODES * FIN * 2);
    float* agg    = (float*)carve((size_t)NNODES * FIN * 4);
    float* cnt    = (float*)carve((size_t)NNODES * 4);
    float* deg    = (float*)carve((size_t)NNODES * 4);
    float* dinv   = (float*)carve((size_t)NNODES * 4);
    float* hgcn   = (float*)carve((size_t)NNODES * HID * 4);
    float* hsr    = (float*)carve((size_t)NNODES * HID * 4);
    float* hsl    = (float*)carve((size_t)NNODES * HID * 4);
    float* hgat   = (float*)carve((size_t)NNODES * HID * 4);
    float* gcnacc = (float*)carve((size_t)NNODES * HID * 4);
    float* gatacc = (float*)carve((size_t)NNODES * HID * 4);
    float* ens    = (float*)carve((size_t)NNODES * HID * 4);
    h16*   ensh   = (h16*)carve((size_t)NNODES * HID * 2);
    float* a_s    = (float*)carve((size_t)NNODES * 4);
    float* a_d    = (float*)carve((size_t)NNODES * 4);
    float* gmax   = (float*)carve((size_t)NNODES * 4);
    float* gsum   = (float*)carve((size_t)NNODES * 4);
    float* qb     = (float*)carve((size_t)NNODES * CPAD * 4);
    float* kb     = (float*)carve((size_t)NNODES * CPAD * 4);
    float* vb     = (float*)carve((size_t)NNODES * CPAD * 4);
    float* sk     = (float*)carve((size_t)NNODES * CPAD * 4);
    float* tout   = (float*)carve((size_t)NNODES * CPAD * 4);
    float* tmax   = (float*)carve((size_t)NNODES * 4);
    float* tsum   = (float*)carve((size_t)NNODES * 4);
    float* tscore = (float*)carve((size_t)NEDGES * 4);
    // packed WMMA B operands (f16, WMMA lane layout)
    h16* pb_gcn = (h16*)carve((size_t)FIN * HID * 2);
    h16* pb_l   = (h16*)carve((size_t)FIN * HID * 2);
    h16* pb_r   = (h16*)carve((size_t)FIN * HID * 2);
    h16* pb_gat = (h16*)carve((size_t)FIN * HID * 2);
    h16* pb_q   = (h16*)carve((size_t)HID * CPAD * 2);
    h16* pb_k   = (h16*)carve((size_t)HID * CPAD * 2);
    h16* pb_v   = (h16*)carve((size_t)HID * CPAD * 2);
    h16* pb_s   = (h16*)carve((size_t)HID * CPAD * 2);

    // zero the accumulators (graph-capturable async memsets)
    hipMemsetAsync(agg,    0, (size_t)NNODES * FIN * 4, stream);
    hipMemsetAsync(cnt,    0, (size_t)NNODES * 4, stream);
    hipMemsetAsync(gcnacc, 0, (size_t)NNODES * HID * 4, stream);
    hipMemsetAsync(gatacc, 0, (size_t)NNODES * HID * 4, stream);
    hipMemsetAsync(gsum,   0, (size_t)NNODES * 4, stream);
    hipMemsetAsync(tsum,   0, (size_t)NNODES * 4, stream);
    hipMemsetAsync(tout,   0, (size_t)NNODES * CPAD * 4, stream);

    k_fill_f32<<<gblk(NNODES), 256, 0, stream>>>(deg, 1.0f, NNODES);       // self loop
    k_fill_f32<<<gblk(NNODES), 256, 0, stream>>>(gmax, -INFINITY, NNODES);
    k_fill_f32<<<gblk(NNODES), 256, 0, stream>>>(tmax, -INFINITY, NNODES);

    // convert x; pack all weights into WMMA B layout (fuses f32->f16 + padding)
    k_f32_to_f16<<<gblk((long long)NNODES * FIN), 256, 0, stream>>>(x, xh, NNODES * FIN);
    k_pack_b<<<gblk(FIN * HID), 256, 0, stream>>>(Wgcn, pb_gcn, FIN, HID, HID / 16);
    k_pack_b<<<gblk(FIN * HID), 256, 0, stream>>>(Wl,   pb_l,   FIN, HID, HID / 16);
    k_pack_b<<<gblk(FIN * HID), 256, 0, stream>>>(Wr,   pb_r,   FIN, HID, HID / 16);
    k_pack_b<<<gblk(FIN * HID), 256, 0, stream>>>(Wgat, pb_gat, FIN, HID, HID / 16);
    k_pack_b<<<gblk(HID * CPAD), 256, 0, stream>>>(Wq, pb_q, HID, NCLS, CPAD / 16);
    k_pack_b<<<gblk(HID * CPAD), 256, 0, stream>>>(Wk, pb_k, HID, NCLS, CPAD / 16);
    k_pack_b<<<gblk(HID * CPAD), 256, 0, stream>>>(Wv, pb_v, HID, NCLS, CPAD / 16);
    k_pack_b<<<gblk(HID * CPAD), 256, 0, stream>>>(Ws, pb_s, HID, NCLS, CPAD / 16);

    // degrees
    k_deg<<<gblk(NEDGES), 256, 0, stream>>>(ei, deg);
    k_dinv<<<gblk(NNODES), 256, 0, stream>>>(deg, dinv);

    // SAGE aggregation
    k_sage_scatter<<<gblk((long long)NEDGES * 32), 256, 0, stream>>>(x, ei, agg, cnt);
    k_sage_norm<<<gblk((long long)NNODES * FIN), 256, 0, stream>>>(agg, cnt, aggh);

    // hidden GEMMs (WMMA): 50000x128 @ 128x64, one wave = 16x64 slab
    const int MT = NNODES / 16;           // 3125 exact
    long long gemmThreads = (long long)MT * 32;
    k_wmma_gemm<4><<<gblk(gemmThreads), 256, 0, stream>>>(xh,   FIN, pb_gcn, hgcn, HID, MT, FIN);
    k_wmma_gemm<4><<<gblk(gemmThreads), 256, 0, stream>>>(xh,   FIN, pb_r,   hsr,  HID, MT, FIN);
    k_wmma_gemm<4><<<gblk(gemmThreads), 256, 0, stream>>>(xh,   FIN, pb_gat, hgat, HID, MT, FIN);
    k_wmma_gemm<4><<<gblk(gemmThreads), 256, 0, stream>>>(aggh, FIN, pb_l,   hsl,  HID, MT, FIN);

    // GCN scatter
    k_gcn_scatter<<<gblk((long long)NEDGES * 32), 256, 0, stream>>>(hgcn, ei, dinv, gcnacc);

    // GAT
    k_gat_pre<<<gblk((long long)NNODES * 32), 256, 0, stream>>>(hgat, asrc, adst, a_s, a_d);
    const int EN = NEDGES + NNODES;
    k_gat_max<<<gblk(EN), 256, 0, stream>>>(ei, a_s, a_d, gmax);
    k_gat_sum<<<gblk(EN), 256, 0, stream>>>(ei, a_s, a_d, gmax, gsum);
    k_gat_scatter<<<gblk((long long)EN * 32), 256, 0, stream>>>(ei, a_s, a_d, gmax, gsum, hgat, gatacc);

    // ensemble
    k_ens<<<gblk((long long)NNODES * HID), 256, 0, stream>>>(
        gcnacc, hgcn, dinv, bgcn, hsl, bl, hsr, gatacc, bgat, w, ens, ensh);

    // transformer projections (WMMA): 50000x64 @ 64x48, one wave = 16x48 slab
    k_wmma_gemm<3><<<gblk(gemmThreads), 256, 0, stream>>>(ensh, HID, pb_q, qb, CPAD, MT, HID);
    k_wmma_gemm<3><<<gblk(gemmThreads), 256, 0, stream>>>(ensh, HID, pb_k, kb, CPAD, MT, HID);
    k_wmma_gemm<3><<<gblk(gemmThreads), 256, 0, stream>>>(ensh, HID, pb_v, vb, CPAD, MT, HID);
    k_wmma_gemm<3><<<gblk(gemmThreads), 256, 0, stream>>>(ensh, HID, pb_s, sk, CPAD, MT, HID);
    k_bias4<<<gblk((long long)NNODES * CPAD), 256, 0, stream>>>(qb, kb, vb, sk, bq, bk, bv, bs);

    // transformer attention
    k_tscore<<<gblk((long long)NEDGES * 32), 256, 0, stream>>>(qb, kb, ei, tscore, tmax);
    k_tsum<<<gblk(NEDGES), 256, 0, stream>>>(ei, tscore, tmax, tsum);
    k_tscatter<<<gblk((long long)NEDGES * 32), 256, 0, stream>>>(ei, tscore, tsum, vb, tout);

    // final skip + log_softmax
    k_final<<<gblk((long long)NNODES * 32), 256, 0, stream>>>(tout, sk, out);
}
